// CustomAttn_90151363543399
// MI455X (gfx1250) — matrix operand order repaired and checked by reading, not compiled, not run on hardware
//
#include <hip/hip_runtime.h>

// ---------------------------------------------------------------------------
// CDNA5 (gfx1250, wave32) WMMA attention pipeline, bf16 datapath end-to-end
//   B=2, S=2048, D=1024, H=16, HD=64
//   qkv = x @ w_in^T ; rope(q,k) ; causal flash attention ; out = o @ w_out^T
// All matmuls via v_wmma_f32_16x16x32_bf16 (f32 accumulate). Flash row-sums
// are computed with a WMMA against an all-ones B fragment instead of lane
// shuffles.
// ---------------------------------------------------------------------------

typedef __attribute__((ext_vector_type(16))) __bf16 bf16x16;
typedef __attribute__((ext_vector_type(8)))  float  v8f;

#define BB 2
#define SS 2048
#define DD 1024
#define HH 16
#define HD 64

static __device__ __forceinline__ v8f wmma_bf16(bf16x16 a, bf16x16 b, v8f c) {
    // (neg_a, A, neg_b, B, c_mod, C, reuse_a, reuse_b)
    return __builtin_amdgcn_wmma_f32_16x16x32_bf16(false, a, false, b,
                                                   (short)0, c, false, false);
}

// 32-byte fragment <-> two 16-byte vector loads
union FragU  { uint4 u[2]; bf16x16 v; };
union Pack4  { __bf16 h[4]; uint2 u; };

// ---------------------------------------------------------------------------
// One-time f32 -> bf16 conversion (n4 = n/4 float4 groups)
// ---------------------------------------------------------------------------
__global__ __launch_bounds__(256)
void cvt_f32_bf16(const float* __restrict__ src, unsigned short* __restrict__ dst,
                  long long n4)
{
    const long long t = (long long)blockIdx.x * blockDim.x + threadIdx.x;
    if (t >= n4) return;
    float4 v = ((const float4*)src)[t];
    Pack4 p;
    p.h[0] = (__bf16)v.x; p.h[1] = (__bf16)v.y;
    p.h[2] = (__bf16)v.z; p.h[3] = (__bf16)v.w;
    ((uint2*)dst)[t] = p.u;
}

// ---------------------------------------------------------------------------
// C[m,n] = sum_k A[m,k] * W[n,k]   (A: MxK bf16, W: NxK bf16, C: MxN f32)
// One wave -> 32x64 output tile (8 WMMA tiles), K-loop step 32.
// Per K-step: 4 b128 loads for A, 8 b128 loads for W, 8 WMMAs, zero cvt.
// The 8 waves of a block land on the same 32-row A tile (consecutive N
// strips), so A fragments broadcast out of L0.
// ---------------------------------------------------------------------------
__global__ __launch_bounds__(256)
void gemm_bf16(const unsigned short* __restrict__ Au,
               const unsigned short* __restrict__ Wu,
               float* __restrict__ C, int M, int N, int K)
{
    const __bf16* A = reinterpret_cast<const __bf16*>(Au);
    const __bf16* W = reinterpret_cast<const __bf16*>(Wu);

    const int lane = threadIdx.x & 31;
    const int wv   = threadIdx.x >> 5;
    const int gw   = blockIdx.x * 8 + wv;
    const int numN = N >> 6;                 // 64-wide N strips
    const int mt   = gw / numN;              // 32-row M tiles
    const int ng   = gw % numN;
    if (mt * 32 >= M) return;                // wave-uniform

    const int nl   = lane & 15;
    const int half = lane >> 4;

    const __bf16* arow0 = A + (size_t)(mt * 32 + nl) * K;
    const __bf16* arow1 = A + (size_t)(mt * 32 + 16 + nl) * K;

    v8f acc[2][4] = {};

    for (int kt = 0; kt < K; kt += 32) {
        // unconditional speculative prefetch of next A K-tile
        __builtin_prefetch(arow0 + kt + 32, 0, 1);
        __builtin_prefetch(arow1 + kt + 32, 0, 1);

        // ---- A fragments: rows mt*32+{nl, 16+nl}; e=0..7 -> K=kt+8h+e,
        //      e=8..15 -> K=kt+16+8h+(e-8)  (both runs contiguous) ----
        FragU af0, af1;
        af0.u[0] = *(const uint4*)(arow0 + kt + 8 * half);
        af0.u[1] = *(const uint4*)(arow0 + kt + 16 + 8 * half);
        af1.u[0] = *(const uint4*)(arow1 + kt + 8 * half);
        af1.u[1] = *(const uint4*)(arow1 + kt + 16 + 8 * half);

        // ---- 4 B fragments: column n = W row n, same K order as A ----
#pragma unroll
        for (int j = 0; j < 4; ++j) {
            const __bf16* wrow =
                W + (size_t)(ng * 64 + j * 16 + nl) * K + kt + 8 * half;
            FragU bf;
            bf.u[0] = *(const uint4*)(wrow);
            bf.u[1] = *(const uint4*)(wrow + 16);
            acc[0][j] = wmma_bf16(af0.v, bf.v, acc[0][j]);
            acc[1][j] = wmma_bf16(af1.v, bf.v, acc[1][j]);
        }
    }

    // ---- store: D[m = r + 8*half][n = nl] per 16x16 tile ----
#pragma unroll
    for (int i = 0; i < 2; ++i)
#pragma unroll
        for (int j = 0; j < 4; ++j)
#pragma unroll
            for (int r = 0; r < 8; ++r)
                C[(size_t)(mt * 32 + i * 16 + r + 8 * half) * N +
                  ng * 64 + j * 16 + nl] = acc[i][j][r];
}

// ---------------------------------------------------------------------------
// RoPE + pack: qkv f32 [B,S,3D] ->
//   Q bf16 [B,H,S,HD]  (pre-scaled by 1/sqrt(HD))
//   K bf16 [B,H,S,HD]
//   Vt bf16 [B,H,HD,S] (transposed so flash PV B-fragments are contiguous)
// ---------------------------------------------------------------------------
__global__ __launch_bounds__(256)
void rope_pack(const float* __restrict__ qkv,
               unsigned short* __restrict__ Qo,
               unsigned short* __restrict__ Ko,
               unsigned short* __restrict__ Vo)
{
    const long long tid = (long long)blockIdx.x * blockDim.x + threadIdx.x;
    if (tid >= (long long)BB * SS * DD) return;

    const int d   = (int)(tid % DD);
    const int row = (int)(tid / DD);       // b*S + s
    const int s   = row % SS;
    const int b   = row / SS;
    const int h   = d >> 6;
    const int i   = d & 63;
    const int j   = i & 31;

    const float* q3 = qkv + (size_t)row * (3 * DD);
    const float qv = q3[d];
    const float kv = q3[DD + d];
    const float vv = q3[2 * DD + d];

    // inv_freq = 10000^(-j/32); angle = s * inv_freq
    const float invf = __expf(-(float)j * (9.210340371976184f / 32.0f));
    const float ang  = (float)s * invf;
    const float c    = __cosf(ang);
    const float sn   = __sinf(ang);

    const int   dp   = (i < 32) ? d + 32 : d - 32;
    const float sgn  = (i < 32) ? -1.0f : 1.0f;

    const float qo = (qv * c + sgn * q3[dp] * sn) * 0.125f;     // fold 1/sqrt(64)
    const float ko =  kv * c + sgn * q3[DD + dp] * sn;

    const size_t oqk = ((size_t)(b * HH + h) * SS + s) * HD + i;
    const size_t ov  = ((size_t)(b * HH + h) * HD + i) * SS + s;
    reinterpret_cast<__bf16*>(Qo)[oqk] = (__bf16)qo;
    reinterpret_cast<__bf16*>(Ko)[oqk] = (__bf16)ko;
    reinterpret_cast<__bf16*>(Vo)[ov]  = (__bf16)vv;
}

// ---------------------------------------------------------------------------
// One 32-key flash chunk: 4 QK WMMAs, online softmax (max via 16-lane
// shuffles, row-sum via WMMA vs all-ones B fragment), 4 PV WMMAs.
// MASKED=true only for the single diagonal chunk.
// ---------------------------------------------------------------------------
template <bool MASKED>
static __device__ __forceinline__
void flash_chunk(const __bf16* __restrict__ Kb, const __bf16* __restrict__ Vtb,
                 const bf16x16 (&qf)[2], const bf16x16& ones,
                 int q0, int kb0, int nl, int half, __bf16 (&ptile)[16][32],
                 v8f (&o_acc)[4], v8f& rsum, float (&rm)[8])
{
    // ---- S = Q K^T : two 16x16 tiles, K-dim 64 ----
    v8f st[2] = {};
#pragma unroll
    for (int t = 0; t < 2; ++t) {
        const __bf16* krow = Kb + (size_t)(kb0 + t * 16 + nl) * HD;
        __builtin_prefetch(krow + 32 * HD, 0, 1);      // next chunk's K rows
#pragma unroll
        for (int f = 0; f < 2; ++f) {
            FragU kf;
            kf.u[0] = *(const uint4*)(krow + f * 32 + 8 * half);
            kf.u[1] = *(const uint4*)(krow + f * 32 + 16 + 8 * half);
            st[t] = wmma_bf16(qf[f], kf.v, st[t]);
        }
    }

    // ---- online softmax (scale already folded into Q) ----
#pragma unroll
    for (int r = 0; r < 8; ++r) {
        float a  = st[0][r];
        float bb = st[1][r];
        if constexpr (MASKED) {
            const int mg = q0 + r + 8 * half;
            a  = (kb0 + nl      <= mg) ? a  : -3.0e38f;
            bb = (kb0 + 16 + nl <= mg) ? bb : -3.0e38f;
        }
        float mx = fmaxf(a, bb);
#pragma unroll
        for (int off = 1; off < 16; off <<= 1)
            mx = fmaxf(mx, __shfl_xor(mx, off, 32));
        const float nm    = fmaxf(rm[r], mx);
        const float alpha = __expf(rm[r] - nm);
        rm[r] = nm;
        ptile[r + 8 * half][nl]      = (__bf16)__expf(a - nm);
        ptile[r + 8 * half][16 + nl] = (__bf16)__expf(bb - nm);
        o_acc[0][r] *= alpha; o_acc[1][r] *= alpha;
        o_acc[2][r] *= alpha; o_acc[3][r] *= alpha;
        rsum[r]     *= alpha;
    }

    // ---- P A-fragment from LDS (16x32 bf16, two contiguous 8-runs) ----
    FragU pf;
    pf.u[0] = *(const uint4*)&ptile[nl][8 * half];
    pf.u[1] = *(const uint4*)&ptile[nl][16 + 8 * half];

    // ---- row sums: rsum[m] += sum_k P[m,k]  (every column of the C tile) ----
    rsum = wmma_bf16(pf.v, ones, rsum);

    // ---- O += P @ V : Vt is [HD, S] so K-dim (keys) is contiguous ----
#pragma unroll
    for (int t = 0; t < 4; ++t) {
        const __bf16* vrow = Vtb + (size_t)(t * 16 + nl) * SS + kb0 + 8 * half;
        __builtin_prefetch(vrow + 32, 0, 1);           // next chunk's V keys
        FragU vf;
        vf.u[0] = *(const uint4*)(vrow);
        vf.u[1] = *(const uint4*)(vrow + 16);
        o_acc[t] = wmma_bf16(pf.v, vf.v, o_acc[t]);
    }
}

// ---------------------------------------------------------------------------
// Causal flash attention. One wave per (b, h, 16-query tile).
// Output written bf16, head-merged [B,S,D] (feeds final GEMM directly).
// ---------------------------------------------------------------------------
__global__ __launch_bounds__(256)
void flash_attn(const unsigned short* __restrict__ Qu,
                const unsigned short* __restrict__ Ku,
                const unsigned short* __restrict__ Vu,
                unsigned short* __restrict__ Ou)
{
    const __bf16* Q  = reinterpret_cast<const __bf16*>(Qu);
    const __bf16* K  = reinterpret_cast<const __bf16*>(Ku);
    const __bf16* Vt = reinterpret_cast<const __bf16*>(Vu);
    __bf16*       O  = reinterpret_cast<__bf16*>(Ou);

    __shared__ __align__(16) __bf16 plds[8][16][32];   // per-wave P tile

    const int lane = threadIdx.x & 31;
    const int wv   = threadIdx.x >> 5;
    const int gw   = blockIdx.x * 8 + wv;
    const int QT   = SS / 16;                          // 128
    const int qt   = gw % QT;
    const int h    = (gw / QT) % HH;
    const int b    = gw / (QT * HH);
    if (b >= BB) return;                               // wave-uniform

    const int q0   = qt * 16;
    const int nl   = lane & 15;
    const int half = lane >> 4;

    const __bf16* Qb  = Q  + (size_t)(b * HH + h) * SS * HD;
    const __bf16* Kb  = K  + (size_t)(b * HH + h) * SS * HD;
    const __bf16* Vtb = Vt + (size_t)(b * HH + h) * HD * SS;

    // ---- Q A-fragments (HD = 64 -> 2 fragments of K=32) ----
    bf16x16 qf[2];
    {
        const __bf16* qrow = Qb + (size_t)(q0 + nl) * HD;
#pragma unroll
        for (int f = 0; f < 2; ++f) {
            FragU t;
            t.u[0] = *(const uint4*)(qrow + f * 32 + 8 * half);
            t.u[1] = *(const uint4*)(qrow + f * 32 + 16 + 8 * half);
            qf[f] = t.v;
        }
    }

    bf16x16 ones;
#pragma unroll
    for (int e = 0; e < 16; ++e) ones[e] = (__bf16)1.0f;

    v8f   o_acc[4] = {};
    v8f   rsum = {};
    float rm[8];
#pragma unroll
    for (int r = 0; r < 8; ++r) rm[r] = -3.0e38f;

    const int nfull = qt / 2;                  // fully-unmasked 32-key chunks
    for (int cck = 0; cck < nfull; ++cck)
        flash_chunk<false>(Kb, Vtb, qf, ones, q0, cck * 32, nl, half,
                           plds[wv], o_acc, rsum, rm);
    // single diagonal (masked) chunk
    flash_chunk<true>(Kb, Vtb, qf, ones, q0, nfull * 32, nl, half,
                      plds[wv], o_acc, rsum, rm);

    // ---- normalize + store bf16, head-merged [B,S,D] ----
#pragma unroll
    for (int t = 0; t < 4; ++t)
#pragma unroll
        for (int r = 0; r < 8; ++r) {
            const int mg = q0 + r + 8 * half;
            O[((size_t)(b * SS + mg)) * DD + h * 64 + t * 16 + nl] =
                (__bf16)(o_acc[t][r] / rsum[r]);
        }
}

// ---------------------------------------------------------------------------
// Host launcher
// ---------------------------------------------------------------------------
extern "C" void kernel_launch(void* const* d_in, const int* in_sizes, int n_in,
                              void* d_out, int out_size, void* d_ws, size_t ws_size,
                              hipStream_t stream)
{
    const float* x     = (const float*)d_in[0];   // [B,S,D]
    const float* w_in  = (const float*)d_in[1];   // [3D,D]
    const float* w_out = (const float*)d_in[2];   // [D,D]
    float*       out   = (float*)d_out;           // [B,S,D]

    char* ws = (char*)d_ws;
    const size_t M = (size_t)BB * SS;             // 4096

    unsigned short* xb   = (unsigned short*)ws;                 // M*D bf16     =  8 MB
    unsigned short* wib  = xb  + M * DD;                        // 3D*D bf16    =  6 MB
    unsigned short* wob  = wib + 3 * DD * DD;                   // D*D bf16     =  2 MB
    float*          qkv  = (float*)(wob + DD * DD);             // M*3D f32     = 48 MB
    unsigned short* Qb   = (unsigned short*)(qkv + M * 3 * DD); //  8 MB
    unsigned short* Kb   = Qb + M * DD;                         //  8 MB
    unsigned short* Vtb  = Kb + M * DD;                         //  8 MB
    unsigned short* Ob   = Vtb + M * DD;                        //  8 MB

    // 0) one-time bf16 packs
    cvt_f32_bf16<<<(int)(M * DD / 4 / 256), 256, 0, stream>>>(x, xb, M * DD / 4);
    cvt_f32_bf16<<<(3 * DD * DD / 4) / 256, 256, 0, stream>>>(w_in, wib, 3 * DD * DD / 4);
    cvt_f32_bf16<<<(DD * DD / 4) / 256, 256, 0, stream>>>(w_out, wob, DD * DD / 4);

    // 1) qkv = x @ w_in^T : M=4096, N=3072, K=1024 -> (128 x 48) waves -> 768 blocks
    gemm_bf16<<<768, 256, 0, stream>>>(xb, wib, qkv, 4096, 3072, 1024);

    // 2) RoPE + pack (Q scaled, V transposed)
    rope_pack<<<(BB * SS * DD) / 256, 256, 0, stream>>>(qkv, Qb, Kb, Vtb);

    // 3) causal flash attention: B*H*(S/16) = 4096 waves -> 512 blocks
    flash_attn<<<512, 256, 0, stream>>>(Qb, Kb, Vtb, Ob);

    // 4) out = O @ w_out^T : M=4096, N=1024, K=1024 -> (128 x 16) waves -> 256 blocks
    gemm_bf16<<<256, 256, 0, stream>>>(Ob, wob, out, 4096, 1024, 1024);
}
